// GATModel_77275051589665
// MI455X (gfx1250) — matrix operand order repaired and checked by reading, not compiled, not run on hardware
//
#include <hip/hip_runtime.h>

#define NN 100000
#define NE 1600000
#define IN_C 128
#define HID_C 64
#define LALPHA 0.2f
#define SEPS 1e-16f

typedef __attribute__((ext_vector_type(16))) __bf16 v16bf;
typedef __attribute__((ext_vector_type(8)))  __bf16 v8bf;
typedef __attribute__((ext_vector_type(8)))  float  v8f;

union AFrag { v16bf v; v8bf h[2]; };

__device__ __forceinline__ unsigned short f32_to_bf16(float f) {
    unsigned u = __float_as_uint(f);
    unsigned r = u + 0x7FFFu + ((u >> 16) & 1u);   // round-to-nearest-even
    return (unsigned short)(r >> 16);
}

// order-preserving float <-> uint mapping for atomicMax on floats
__device__ __forceinline__ unsigned f2ord(float f) {
    unsigned u = __float_as_uint(f);
    return (u & 0x80000000u) ? ~u : (u | 0x80000000u);
}
__device__ __forceinline__ float ord2f(unsigned k) {
    unsigned u = (k & 0x80000000u) ? (k ^ 0x80000000u) : ~k;
    return __uint_as_float(u);
}

// ---------------------------------------------------------------------------
// GEMM: out[M,64] = X[M,K] @ W[K,64] + bias   (bf16 WMMA, f32 accumulate)
// block = 256 threads = 8 waves; block tile M=128, N=64; K step 32
// ---------------------------------------------------------------------------
__global__ __launch_bounds__(256) void gemm_bias(
    const float* __restrict__ X, const float* __restrict__ W,
    const float* __restrict__ bias, float* __restrict__ out, int M, int K) {
    __shared__ alignas(64) unsigned short Xs[128 * 32];   // 8 KB
    __shared__ alignas(64) unsigned short Wt[64 * 128];   // 16 KB, Wt[n*K + k]

    const int tid  = threadIdx.x;
    const int lane = tid & 31;
    const int wave = tid >> 5;
    const long blockM = (long)blockIdx.x * 128;

    // preload W transposed to bf16
    for (int e = tid; e < K * 64; e += 256) {
        int k = e >> 6, n = e & 63;
        Wt[n * K + k] = f32_to_bf16(W[k * 64 + n]);
    }
    __syncthreads();

    v8f acc[4] = {};

    const int arow   = wave * 16 + (lane & 15);      // local A row
    const int c0     = (lane & 16) ? 16 : 0;         // A K-group base per lane half
    const int khalfB = (lane & 16) ? 16 : 0;         // B K base per lane half
    const int ncol   = lane & 15;

    for (int kt = 0; kt < K; kt += 32) {
        // stage 128x32 bf16 tile of X
        {
            int row  = tid >> 1;
            int half = (tid & 1) * 16;
            long gr = blockM + row; if (gr >= M) gr = M - 1;
            const float* src = X + gr * (long)K + kt + half;
            #pragma unroll
            for (int q = 0; q < 4; ++q) {
                float4 f = *(const float4*)(src + q * 4);
                int base = row * 32 + half + q * 4;
                Xs[base + 0] = f32_to_bf16(f.x);
                Xs[base + 1] = f32_to_bf16(f.y);
                Xs[base + 2] = f32_to_bf16(f.z);
                Xs[base + 3] = f32_to_bf16(f.w);
            }
        }
        __syncthreads();

        AFrag a;
        a.h[0] = *(const v8bf*)&Xs[arow * 32 + c0];
        a.h[1] = *(const v8bf*)&Xs[arow * 32 + c0 + 8];
        #pragma unroll
        for (int nt = 0; nt < 4; ++nt) {
            int n = nt * 16 + ncol;
            v16bf b = *(const v16bf*)&Wt[n * K + kt + khalfB];
            acc[nt] = __builtin_amdgcn_wmma_f32_16x16x32_bf16(
                false, a.v, false, b, (short)0, acc[nt], false, false);
        }
        __syncthreads();
    }

    // epilogue: C layout — VGPR j: row = base + j + (lane>=16 ? 8:0), col = n0 + (lane&15)
    const int rAdd = (lane & 16) ? 8 : 0;
    const long rowBase = blockM + wave * 16 + rAdd;
    if (blockM + 128 <= M) {
        // full tile: straight-line coalesced stores, no per-element guard
        #pragma unroll
        for (int nt = 0; nt < 4; ++nt) {
            int col = nt * 16 + ncol;
            float bv = bias[col];
            float* op = out + rowBase * 64 + col;
            #pragma unroll
            for (int j = 0; j < 8; ++j)
                op[j * 64] = acc[nt][j] + bv;
        }
    } else {
        #pragma unroll
        for (int nt = 0; nt < 4; ++nt) {
            int col = nt * 16 + ncol;
            float bv = bias[col];
            #pragma unroll
            for (int j = 0; j < 8; ++j) {
                long row = rowBase + j;
                if (row < M) out[row * 64 + col] = acc[nt][j] + bv;
            }
        }
    }
}

// ---------------------------------------------------------------------------
// per-node attention scores: s_i = Wh . A[:64], s_j = Wh . A[64:128]
// one wave32 per node
// ---------------------------------------------------------------------------
__global__ __launch_bounds__(256) void node_scores(
    const float* __restrict__ Wh, const float* __restrict__ A,
    float* __restrict__ si, float* __restrict__ sj, int n) {
    int wave = threadIdx.x >> 5;
    int lane = threadIdx.x & 31;
    int node = blockIdx.x * 8 + wave;
    if (node >= n) return;
    float h0 = Wh[(long)node * 64 + lane];
    float h1 = Wh[(long)node * 64 + 32 + lane];
    float ai = h0 * A[lane] + h1 * A[32 + lane];
    float aj = h0 * A[64 + lane] + h1 * A[96 + lane];
    #pragma unroll
    for (int off = 16; off >= 1; off >>= 1) {
        ai += __shfl_xor(ai, off, 32);
        aj += __shfl_xor(aj, off, 32);
    }
    if (lane == 0) { si[node] = ai; sj[node] = aj; }
}

// pass 1: segment max via ordered-uint atomicMax
__global__ __launch_bounds__(256) void edge_max(
    const int* __restrict__ ei, const float* __restrict__ si,
    const float* __restrict__ sj, const float* __restrict__ bA,
    unsigned* __restrict__ mu, int nE) {
    int e = blockIdx.x * 256 + threadIdx.x;
    if (e >= nE) return;
    int src = ei[e], dst = ei[nE + e];
    float v = si[dst] + sj[src] + bA[0];
    v = v >= 0.f ? v : LALPHA * v;
    atomicMax(&mu[dst], f2ord(v));
}

// pass 2: ex = exp(e - m), denom += ex ; prefetch Wh rows for pass 3
__global__ __launch_bounds__(256) void edge_exp(
    const int* __restrict__ ei, const float* __restrict__ si,
    const float* __restrict__ sj, const float* __restrict__ bA,
    const unsigned* __restrict__ mu, float* __restrict__ exv,
    float* __restrict__ denom, const float* __restrict__ Wh, int nE) {
    int e = blockIdx.x * 256 + threadIdx.x;
    if (e >= nE) return;
    int src = ei[e], dst = ei[nE + e];
    float v = si[dst] + sj[src] + bA[0];
    v = v >= 0.f ? v : LALPHA * v;
    float m = ord2f(mu[dst]);
    float x = __expf(v - m);
    exv[e] = x;
    atomicAdd(&denom[dst], x);
    __builtin_prefetch(&Wh[(long)src * 64], 0, 1);  // warm L2 for scatter pass
}

// pass 3: out[dst] += (ex/denom) * Wh[src] ; one wave per edge, float2/lane
__global__ __launch_bounds__(256) void edge_scatter(
    const int* __restrict__ ei, const float* __restrict__ exv,
    const float* __restrict__ denom, const float* __restrict__ Wh,
    float* __restrict__ out, int nE) {
    long t = (long)blockIdx.x * 256 + threadIdx.x;
    long e = t >> 5;
    if (e >= nE) return;
    int lane = threadIdx.x & 31;
    int src = ei[e], dst = ei[nE + e];
    float al = exv[e] / (denom[dst] + SEPS);
    float2 w = *(const float2*)(Wh + (long)src * 64 + lane * 2);
    float* o = out + (long)dst * 64 + lane * 2;
    atomicAdd(o, al * w.x);
    atomicAdd(o + 1, al * w.y);
}

__global__ __launch_bounds__(256) void leaky_inplace(float* __restrict__ v, long n) {
    long i = (long)blockIdx.x * 256 + threadIdx.x;
    if (i < n) { float x = v[i]; v[i] = x >= 0.f ? x : LALPHA * x; }
}

// ---------------------------------------------------------------------------
extern "C" void kernel_launch(void* const* d_in, const int* in_sizes, int n_in,
                              void* d_out, int out_size, void* d_ws, size_t ws_size,
                              hipStream_t stream) {
    (void)in_sizes; (void)n_in; (void)out_size; (void)ws_size;
    const float* x   = (const float*)d_in[0];
    const int*   ei  = (const int*)d_in[1];
    const float* W1  = (const float*)d_in[2];
    const float* bW1 = (const float*)d_in[3];
    const float* A1  = (const float*)d_in[4];
    const float* bA1 = (const float*)d_in[5];
    const float* W2  = (const float*)d_in[6];
    const float* bW2 = (const float*)d_in[7];
    const float* A2  = (const float*)d_in[8];
    const float* bA2 = (const float*)d_in[9];
    const float* Wfc = (const float*)d_in[10];
    const float* bfc = (const float*)d_in[11];
    float* out = (float*)d_out;

    float* ws = (float*)d_ws;
    float*    Wh    = ws;                              // NN*64
    float*    agg   = Wh + (size_t)NN * 64;            // NN*64
    float*    si    = agg + (size_t)NN * 64;           // NN
    float*    sj    = si + NN;                         // NN
    unsigned* mu    = (unsigned*)(sj + NN);            // NN
    float*    denom = (float*)(mu + NN);               // NN
    float*    exv   = denom + NN;                      // NE

    const long nh = (long)NN * 64;
    dim3 B(256);
    int gGemm = (NN + 127) / 128;
    int gNode = (NN + 7) / 8;
    int gEdge = (NE + 255) / 256;
    int gScat = (int)(((long)NE * 32 + 255) / 256);
    int gElem = (int)((nh + 255) / 256);

    // ----- layer 1 -----
    gemm_bias<<<gGemm, B, 0, stream>>>(x, W1, bW1, Wh, NN, IN_C);
    node_scores<<<gNode, B, 0, stream>>>(Wh, A1, si, sj, NN);
    hipMemsetAsync(mu, 0, (size_t)NN * 4, stream);
    hipMemsetAsync(denom, 0, (size_t)NN * 4, stream);
    hipMemsetAsync(agg, 0, (size_t)nh * 4, stream);
    edge_max<<<gEdge, B, 0, stream>>>(ei, si, sj, bA1, mu, NE);
    edge_exp<<<gEdge, B, 0, stream>>>(ei, si, sj, bA1, mu, exv, denom, Wh, NE);
    edge_scatter<<<gScat, B, 0, stream>>>(ei, exv, denom, Wh, agg, NE);
    leaky_inplace<<<gElem, B, 0, stream>>>(agg, nh);

    // ----- layer 2 -----
    gemm_bias<<<gGemm, B, 0, stream>>>(agg, W2, bW2, Wh, NN, HID_C);
    node_scores<<<gNode, B, 0, stream>>>(Wh, A2, si, sj, NN);
    hipMemsetAsync(mu, 0, (size_t)NN * 4, stream);
    hipMemsetAsync(denom, 0, (size_t)NN * 4, stream);
    hipMemsetAsync(agg, 0, (size_t)nh * 4, stream);
    edge_max<<<gEdge, B, 0, stream>>>(ei, si, sj, bA2, mu, NE);
    edge_exp<<<gEdge, B, 0, stream>>>(ei, si, sj, bA2, mu, exv, denom, Wh, NE);
    edge_scatter<<<gScat, B, 0, stream>>>(ei, exv, denom, Wh, agg, NE);
    leaky_inplace<<<gElem, B, 0, stream>>>(agg, nh);

    // ----- final FC -----
    gemm_bias<<<gGemm, B, 0, stream>>>(agg, Wfc, bfc, out, NN, HID_C);
}
